// LSMPool_58110907515598
// MI455X (gfx1250) — matrix operand order repaired and checked by reading, not compiled, not run on hardware
//
#include <hip/hip_runtime.h>
#include <cstdint>
#include <cstddef>

// ---------------------------------------------------------------------------
// Spiking LSM (reservoir + readout) on MI455X / gfx1250.
//
// Per timestep t (serial recurrence forces per-step kernels):
//   K1: cur = [x_t | spk_prev] @ w1^T   (64x5120 -> 64x4096), fused LinearLeakLIF
//   K2: cur2 = spk_h @ w2^T + b2        (64x4096 -> 64x512),  fused LIF -> out[t]
//
// Precision: w = hi(bf16) + lo(bf16); spikes are exact in bf16, so
// C += A*Bhi + A*Blo with f32 accumulate ~ reproduces fp32 GEMM.
// Matrix math via v_wmma_f32_16x16x32_bf16 (wave32 CDNA5 WMMA).
// K is split 4-ways across the waves of a block; partials reduced in LDS.
// ---------------------------------------------------------------------------

typedef __bf16 bf16;
typedef __attribute__((ext_vector_type(16))) __bf16 v16bf;
typedef __attribute__((ext_vector_type(8)))  float  v8f;

#define T_STEPS 128
#define BATCH   64
#define IN_F    1024
#define HID_F   4096
#define OUT_F   512
#define TOTAL_F (IN_F + HID_F)   // 5120

// Workspace byte offsets
#define OFF_W1HI 0ULL
#define OFF_W1LO 41943040ULL     // 4096*5120*2
#define OFF_W2HI 83886080ULL
#define OFF_W2LO 88080384ULL     // + 512*4096*2
#define OFF_XSW  92274688ULL
#define OFF_SPK0 109051904ULL    // + 128*64*1024*2
#define OFF_SPK1 109576192ULL    // + 64*4096*2
#define OFF_MEM1 110100480ULL
#define OFF_MEM2 111149056ULL    // + 64*4096*4
#define WS_END   111280128ULL    // + 64*512*4

// ---- fragment index helpers (CDNA5 16-bit WMMA VGPR layouts, ISA 7.12.2) ----

// A (16x32 bf16): lane = ((k%32/8)&1)*16 + m%16 ; j = k%8 + (k%32>=16 ? 8:0)
// tiles stored as tile_id = kt*4 + mt, 512 bf16 per tile, lane-contiguous.
__device__ __forceinline__ size_t afrag_off(int m, int k) {
  int kt = k >> 5, kl = k & 31;
  int lane = ((kl >> 3) & 1) * 16 + (m & 15);
  int j = (kl & 7) + (kl >= 16 ? 8 : 0);
  return ((size_t)(kt * 4 + (m >> 4))) * 512 + (size_t)lane * 16 + j;
}

// ---------------------------- init / prepack --------------------------------

__global__ void zero_words_kernel(uint32_t* __restrict__ p, unsigned n) {
  unsigned i = blockIdx.x * blockDim.x + threadIdx.x;
  if (i < n) p[i] = 0u;
}

// w: (N,K) row-major fp32; B-fragment (k,n): lane = ((k%32/8)&1)*16 + n%16, j = k%8 + (k%32>=16 ? 8:0)
// tiles stored tile_id = nt*Kt + kt (K-contiguous per 16-col group).
__global__ void prepack_w_kernel(const float* __restrict__ w,
                                 bf16* __restrict__ hi, bf16* __restrict__ lo,
                                 int N, int K) {
  int id = blockIdx.x * blockDim.x + threadIdx.x;
  if (id >= N * K) return;
  int k = id % K, n = id / K;
  float v = w[id];
  bf16 h = (bf16)v;
  bf16 l = (bf16)(v - (float)h);
  int Kt = K >> 5;
  int kt = k >> 5, kl = k & 31;
  int lane = ((kl >> 3) & 1) * 16 + (n & 15);
  int j = (kl & 7) + (kl >= 16 ? 8 : 0);
  size_t off = ((size_t)(n >> 4) * Kt + kt) * 512 + (size_t)lane * 16 + j;
  hi[off] = h;
  lo[off] = l;
}

// x: (T,B,IN) fp32 spikes (0/1) -> per-timestep A-fragment layout (bf16 exact)
__global__ void prepack_x_kernel(const float* __restrict__ x, bf16* __restrict__ dst) {
  int id = blockIdx.x * blockDim.x + threadIdx.x;
  if (id >= T_STEPS * BATCH * IN_F) return;
  int k = id % IN_F;
  int m = (id / IN_F) % BATCH;
  int t = id / (IN_F * BATCH);
  dst[(size_t)t * (BATCH * IN_F) + afrag_off(m, k)] = (bf16)x[id];
}

// ------------------------ timestep kernels ----------------------------------

#define WMMA_BF16(A, B, C) \
  __builtin_amdgcn_wmma_f32_16x16x32_bf16(false, (A), false, (B), (short)0, (C), false, false)

// One 32-K-step of the hi/lo GEMM: 4 A-frags x {Bhi,Blo} -> 8 WMMAs.
#define GEMM_STEP(APTR, BH, BL)                                   \
  {                                                               \
    v16bf a0 = *(const v16bf*)((APTR));                           \
    v16bf a1 = *(const v16bf*)((APTR) + 512);                     \
    v16bf a2 = *(const v16bf*)((APTR) + 1024);                    \
    v16bf a3 = *(const v16bf*)((APTR) + 1536);                    \
    v16bf bh = *(const v16bf*)(BH);                               \
    v16bf bl = *(const v16bf*)(BL);                               \
    c0 = WMMA_BF16(a0, bh, c0);                                   \
    c1 = WMMA_BF16(a1, bh, c1);                                   \
    c2 = WMMA_BF16(a2, bh, c2);                                   \
    c3 = WMMA_BF16(a3, bh, c3);                                   \
    c0 = WMMA_BF16(a0, bl, c0);                                   \
    c1 = WMMA_BF16(a1, bl, c1);                                   \
    c2 = WMMA_BF16(a2, bl, c2);                                   \
    c3 = WMMA_BF16(a3, bl, c3);                                   \
  }

// GEMM1 (64x4096, K=5120) + LinearLeakLIF(relu) fused.
// grid 256 x 128 threads: block = one 16-col group; 4 waves split K (40 kt each);
// LDS reduction, then wave w retires M-tile mt==w.
__global__ __launch_bounds__(128)
void lsm_step_kernel(int t,
                     const bf16* __restrict__ xsw,
                     const bf16* __restrict__ spkPrev,
                     bf16* __restrict__ spkNext,
                     const bf16* __restrict__ w1hi,
                     const bf16* __restrict__ w1lo,
                     const float* __restrict__ beta,
                     const float* __restrict__ thr,
                     float* __restrict__ mem1) {
  __shared__ float red[4 * 4 * 32 * 8];   // [wave][mt][lane][r], 16 KB
  const int lane = threadIdx.x & 31;
  const int wv   = threadIdx.x >> 5;      // K-split index 0..3
  const int ng   = blockIdx.x;            // column group 0..255
  const bf16* xs  = xsw + (size_t)t * (BATCH * IN_F);
  const bf16* bph = w1hi + ((size_t)ng * 160) * 512 + (size_t)lane * 16;
  const bf16* bpl = w1lo + ((size_t)ng * 160) * 512 + (size_t)lane * 16;

  const int k0 = wv * 40, k1 = k0 + 40;
  v8f c0 = {}, c1 = {}, c2 = {}, c3 = {};

  // phase 1: input spikes x_t (kt < 32)
  const int xEnd = (k1 < 32) ? k1 : 32;
  for (int kt = k0; kt < xEnd; ++kt) {
    const bf16* ap = xs + (size_t)(kt * 4) * 512 + (size_t)lane * 16;
    GEMM_STEP(ap, bph + (size_t)kt * 512, bpl + (size_t)kt * 512);
  }
  // phase 2: recurrent spikes (kt >= 32)
  const int rBeg = (k0 > 32) ? k0 : 32;
  for (int kt = rBeg; kt < k1; ++kt) {
    const bf16* ap = spkPrev + (size_t)((kt - 32) * 4) * 512 + (size_t)lane * 16;
    GEMM_STEP(ap, bph + (size_t)kt * 512, bpl + (size_t)kt * 512);
  }

  // cross-wave K reduction in LDS
  *(v8f*)&red[((wv * 4 + 0) * 32 + lane) * 8] = c0;
  *(v8f*)&red[((wv * 4 + 1) * 32 + lane) * 8] = c1;
  *(v8f*)&red[((wv * 4 + 2) * 32 + lane) * 8] = c2;
  *(v8f*)&red[((wv * 4 + 3) * 32 + lane) * 8] = c3;
  __syncthreads();
  v8f acc = {};
#pragma unroll
  for (int p = 0; p < 4; ++p)
    acc = acc + *(const v8f*)&red[((p * 4 + wv) * 32 + lane) * 8];

  // fused LIF for M-tile mt = wv: mem = relu(mem - beta + cur); spike; subtract
  const int n = (ng << 4) + (lane & 15);
  const float be = beta[n], th = thr[n];
  const int mhalf = (lane < 16) ? 0 : 8;
#pragma unroll
  for (int r = 0; r < 8; ++r) {
    int m = (wv << 4) + r + mhalf;
    size_t idx = (size_t)m * HID_F + n;
    float mem = mem1[idx] - be + acc[r];
    mem = fmaxf(mem, 0.0f);
    float s = (mem > th) ? 1.0f : 0.0f;
    mem -= s * th;
    mem1[idx] = mem;
    spkNext[afrag_off(m, n)] = (bf16)s;   // exact 0/1 in bf16
  }
}

// GEMM2 (64x512, K=4096) + LIF (no relu) fused; writes out[t] spikes.
// grid 32 x 128 threads: block = one 16-col group; 4 waves split K (32 kt each).
__global__ __launch_bounds__(128)
void readout_step_kernel(int t,
                         const bf16* __restrict__ spk,
                         const bf16* __restrict__ w2hi,
                         const bf16* __restrict__ w2lo,
                         const float* __restrict__ b2,
                         const float* __restrict__ beta,
                         const float* __restrict__ thr,
                         float* __restrict__ mem2,
                         float* __restrict__ out) {
  __shared__ float red[4 * 4 * 32 * 8];   // 16 KB
  const int lane = threadIdx.x & 31;
  const int wv   = threadIdx.x >> 5;      // K-split index 0..3
  const int ng   = blockIdx.x;            // column group 0..31
  const bf16* bph = w2hi + ((size_t)ng * 128) * 512 + (size_t)lane * 16;
  const bf16* bpl = w2lo + ((size_t)ng * 128) * 512 + (size_t)lane * 16;

  const int k0 = wv * 32, k1 = k0 + 32;
  v8f c0 = {}, c1 = {}, c2 = {}, c3 = {};
  for (int kt = k0; kt < k1; ++kt) {
    const bf16* ap = spk + (size_t)(kt * 4) * 512 + (size_t)lane * 16;
    GEMM_STEP(ap, bph + (size_t)kt * 512, bpl + (size_t)kt * 512);
  }

  *(v8f*)&red[((wv * 4 + 0) * 32 + lane) * 8] = c0;
  *(v8f*)&red[((wv * 4 + 1) * 32 + lane) * 8] = c1;
  *(v8f*)&red[((wv * 4 + 2) * 32 + lane) * 8] = c2;
  *(v8f*)&red[((wv * 4 + 3) * 32 + lane) * 8] = c3;
  __syncthreads();
  v8f acc = {};
#pragma unroll
  for (int p = 0; p < 4; ++p)
    acc = acc + *(const v8f*)&red[((p * 4 + wv) * 32 + lane) * 8];

  const int n = (ng << 4) + (lane & 15);
  const float bias = b2[n], be = beta[n], th = thr[n];
  const int mhalf = (lane < 16) ? 0 : 8;
  float* outT = out + (size_t)t * BATCH * OUT_F;
#pragma unroll
  for (int r = 0; r < 8; ++r) {
    int m = (wv << 4) + r + mhalf;
    size_t idx = (size_t)m * OUT_F + n;
    float mem = mem2[idx] - be + (acc[r] + bias);
    float s = (mem > th) ? 1.0f : 0.0f;
    mem -= s * th;
    mem2[idx] = mem;
    outT[idx] = s;
  }
}

// ------------------------------- launcher -----------------------------------

extern "C" void kernel_launch(void* const* d_in, const int* in_sizes, int n_in,
                              void* d_out, int out_size, void* d_ws, size_t ws_size,
                              hipStream_t stream) {
  const float* x        = (const float*)d_in[0];
  const float* w1       = (const float*)d_in[1];
  const float* w2       = (const float*)d_in[2];
  const float* b2       = (const float*)d_in[3];
  const float* beta_lsm = (const float*)d_in[4];
  const float* thr_lsm  = (const float*)d_in[5];
  const float* beta_ro  = (const float*)d_in[6];
  const float* thr_ro   = (const float*)d_in[7];
  float* out = (float*)d_out;

  char* ws   = (char*)d_ws;
  bf16*  w1hi = (bf16*)(ws + OFF_W1HI);
  bf16*  w1lo = (bf16*)(ws + OFF_W1LO);
  bf16*  w2hi = (bf16*)(ws + OFF_W2HI);
  bf16*  w2lo = (bf16*)(ws + OFF_W2LO);
  bf16*  xsw  = (bf16*)(ws + OFF_XSW);
  bf16*  spk0 = (bf16*)(ws + OFF_SPK0);
  bf16*  spk1 = (bf16*)(ws + OFF_SPK1);
  float* mem1 = (float*)(ws + OFF_MEM1);
  float* mem2 = (float*)(ws + OFF_MEM2);

  // zero dynamic state (spike ping/pong + membranes) — contiguous region
  {
    unsigned nwords = (unsigned)((WS_END - OFF_SPK0) / 4);
    zero_words_kernel<<<(nwords + 255) / 256, 256, 0, stream>>>(
        (uint32_t*)(ws + OFF_SPK0), nwords);
  }

  // prepack weights (hi/lo bf16, B-fragment swizzle) and inputs (A-fragment)
  prepack_w_kernel<<<(HID_F * TOTAL_F + 255) / 256, 256, 0, stream>>>(
      w1, w1hi, w1lo, HID_F, TOTAL_F);
  prepack_w_kernel<<<(OUT_F * HID_F + 255) / 256, 256, 0, stream>>>(
      w2, w2hi, w2lo, OUT_F, HID_F);
  prepack_x_kernel<<<(T_STEPS * BATCH * IN_F + 255) / 256, 256, 0, stream>>>(x, xsw);

  // serial recurrence: 2 fused kernels per timestep, ping-pong spike buffers
  for (int t = 0; t < T_STEPS; ++t) {
    bf16* prev = (t & 1) ? spk1 : spk0;
    bf16* next = (t & 1) ? spk0 : spk1;
    lsm_step_kernel<<<256, 128, 0, stream>>>(t, xsw, prev, next, w1hi, w1lo,
                                             beta_lsm, thr_lsm, mem1);
    readout_step_kernel<<<32, 128, 0, stream>>>(t, next, w2hi, w2lo, b2,
                                                beta_ro, thr_ro, mem2, out);
  }
}
